// MultiHeadAttention_19963007992266
// MI455X (gfx1250) — matrix operand order repaired and checked by reading, not compiled, not run on hardware
//
#include <hip/hip_runtime.h>
#include <hip/hip_bf16.h>
#include <stdint.h>

typedef __bf16 bf16;
typedef __attribute__((ext_vector_type(16))) bf16  v16bf;
typedef __attribute__((ext_vector_type(8)))  bf16  v8bf;
typedef __attribute__((ext_vector_type(8)))  float v8f;

#define B_ 4
#define T_ 2048
#define C_ 1024
#define H_ 16
#define D_ 64
#define M_ (B_ * T_)   // 8192 rows

// ---------- WMMA fragment helpers (layouts per CDNA5 ISA 7.12.2 / 7.12.4) ----------

// A-matrix 16x32 bf16 from row-major src (ld elems/row), p -> element (0,0).
// lane L: m = L&15, half = L>>4; elems 0..7 -> K = half*8+i ; elems 8..15 -> K = 16+half*8+(i-8)
__device__ __forceinline__ v16bf load_a16x32(const bf16* p, int ld) {
    const int lane = threadIdx.x & 31;
    const int m = lane & 15, hf = lane >> 4;
    union { v16bf v; v8bf h[2]; } u;
    u.h[0] = *(const v8bf*)(p + (size_t)m * ld + hf * 8);
    u.h[1] = *(const v8bf*)(p + (size_t)m * ld + hf * 8 + 16);
    return u.v;
}

// B-matrix 32x16 bf16 where B[k][n] = W[n0+n][k0+k], W row-major (ld), p -> W[n0][k0].
// lane L: n = L&15, half = L>>4; elems 0..7 -> K = half*8+i ; elems 8..15 -> K = 16+half*8+(i-8)
__device__ __forceinline__ v16bf load_b32x16(const bf16* p, int ld) {
    const int lane = threadIdx.x & 31;
    const int n = lane & 15, hf = lane >> 4;
    union { v16bf v; v8bf h[2]; } u;
    u.h[0] = *(const v8bf*)(p + (size_t)n * ld + hf * 8);
    u.h[1] = *(const v8bf*)(p + (size_t)n * ld + hf * 8 + 16);
    return u.v;
}

__device__ __forceinline__ v8f wmma_bf16(v16bf a, v16bf b, v8f c) {
    return __builtin_amdgcn_wmma_f32_16x16x32_bf16(
        false, a, false, b, (short)0, c, false, false);
}

__device__ __forceinline__ v8f vzero8() {
    v8f z = {0.f, 0.f, 0.f, 0.f, 0.f, 0.f, 0.f, 0.f};
    return z;
}

#define WAIT_ASYNC()    asm volatile("s_wait_asynccnt 0x0" ::: "memory")
#define WAIT_DS()       asm volatile("s_wait_dscnt 0x0" ::: "memory")
// Raw split barrier: no implicit loadcnt/storecnt drain (keeps global prefetches in flight).
// LDS producer/consumer safety: async stores waited via WAIT_ASYNC before signaling, and
// B-frag ds_loads were already dscnt-waited before their consuming WMMAs.
#define BLOCK_BARRIER() asm volatile("s_barrier_signal -1\n\ts_barrier_wait -1" ::: "memory")

// Async-copy 32 bytes/thread of a 64(n) x 32(k) bf16 B tile into LDS (4 KB total, 128 threads).
// LDS layout: row n is 32 consecutive bf16 (64 B), so load_b32x16(lds + 16*t*32, 32) reads frags.
// Instruction offset applies to BOTH the LDS and global address (ISA async pseudocode).
__device__ __forceinline__ void stage_b_tile_async(const bf16* wCol, int k0, bf16* dstLds) {
    const int tid  = threadIdx.x;       // 0..127
    const int srow = tid >> 1;          // 0..63 : n within tile
    const int shalf = tid & 1;          // 0..1  : 32-byte half of the 64-byte row
    const unsigned long long ga =
        (unsigned long long)(uintptr_t)(wCol + (size_t)srow * C_ + k0 + shalf * 16);
    const unsigned lo =
        (unsigned)(uintptr_t)(dstLds + srow * 32 + shalf * 16);  // LDS byte offset = addr[31:0]
    asm volatile("global_load_async_to_lds_b128 %0, %1, off"           :: "v"(lo), "v"(ga) : "memory");
    asm volatile("global_load_async_to_lds_b128 %0, %1, off offset:16" :: "v"(lo), "v"(ga) : "memory");
}

// Cooperative 128x64 block-tile GEMM mainloop: 4 waves, each 32 rows; B tile shared via
// double-buffered LDS staged with async-to-LDS; A frags and B stage software-pipelined.
// unroll 1 + pointer swap keeps accumulators in-place (D==C register coalescing).
__device__ __forceinline__ void coop_gemm_mainloop(
    const bf16* __restrict__ aRow,   // A at (waveRow0, 0), row-major ld=C_
    const bf16* __restrict__ wCol,   // W at (col0, 0),    row-major ld=C_
    bf16* lds0, bf16* lds1,          // two 64x32 bf16 buffers
    v8f acc[2][4])
{
    bf16* cur = lds0;
    bf16* nxt = lds1;
    stage_b_tile_async(wCol, 0, cur);
    v16bf a0 = load_a16x32(aRow, C_);
    v16bf a1 = load_a16x32(aRow + 16 * C_, C_);
    WAIT_ASYNC();
    BLOCK_BARRIER();

#pragma unroll 1
    for (int i = 0; i < C_ / 32 - 1; ++i) {
        const int k0 = i * 32;
        // prefetch next step before consuming current fragments
        stage_b_tile_async(wCol, k0 + 32, nxt);
        v16bf a0n = load_a16x32(aRow + k0 + 32, C_);
        v16bf a1n = load_a16x32(aRow + 16 * C_ + k0 + 32, C_);
#pragma unroll
        for (int t = 0; t < 4; ++t) {
            v16bf b = load_b32x16(cur + 16 * t * 32, 32);
            acc[0][t] = wmma_bf16(a0, b, acc[0][t]);
            acc[1][t] = wmma_bf16(a1, b, acc[1][t]);
        }
        WAIT_ASYNC();            // own async stores to LDS complete
        BLOCK_BARRIER();         // all waves staged; prior reads already register-resident
        bf16* tmp = cur; cur = nxt; nxt = tmp;
        a0 = a0n; a1 = a1n;
    }
    // peeled last K-step
#pragma unroll
    for (int t = 0; t < 4; ++t) {
        v16bf b = load_b32x16(cur + 16 * t * 32, 32);
        acc[0][t] = wmma_bf16(a0, b, acc[0][t]);
        acc[1][t] = wmma_bf16(a1, b, acc[1][t]);
    }
}

// ---------- f32 -> bf16 conversion ----------
__global__ void cvt_f32_bf16(const float* __restrict__ in, bf16* __restrict__ out, int n) {
    int i = blockIdx.x * blockDim.x + threadIdx.x;
    if (i < n) out[i] = (bf16)in[i];
}

// ---------- QKV projection: qkv = x @ w_qkv^T, scattered into q,k [B,H,T,D], v^T [B,H,D,T] ----------
__global__ __launch_bounds__(128) void qkv_gemm_wmma(
    const bf16* __restrict__ xb,    // [M, C]
    const bf16* __restrict__ wb,    // [3C, C]
    bf16* __restrict__ qb,          // [B*H, T, D]
    bf16* __restrict__ kb,          // [B*H, T, D]
    bf16* __restrict__ vt)          // [B*H, D, T]
{
    __shared__ __attribute__((aligned(128))) bf16 btile[2][64 * 32];
    const int lane = threadIdx.x & 31;
    const int wid  = threadIdx.x >> 5;
    const int blockRow0 = (blockIdx.x / 48) * 128;   // M/128 x 3072/64 = 64 x 48 blocks
    const int col0      = (blockIdx.x % 48) * 64;
    const int waveRow0  = blockRow0 + wid * 32;

    v8f acc[2][4] = {{vzero8(), vzero8(), vzero8(), vzero8()},
                     {vzero8(), vzero8(), vzero8(), vzero8()}};
    coop_gemm_mainloop(xb + (size_t)waveRow0 * C_, wb + (size_t)col0 * C_,
                       &btile[0][0], &btile[1][0], acc);

    const int n = lane & 15, hf = lane >> 4;
#pragma unroll
    for (int u = 0; u < 2; ++u) {
#pragma unroll
        for (int t = 0; t < 4; ++t) {
            const int gn = col0 + 16 * t + n;
            const int which = gn >> 10;     // 0=q 1=k 2=v
            const int c = gn & 1023;
            const int hh = c >> 6, dd = c & 63;
#pragma unroll
            for (int r = 0; r < 8; ++r) {
                const int gm = waveRow0 + 16 * u + r + 8 * hf;   // global row = b*T + t
                const int bb = gm >> 11, tt = gm & (T_ - 1);
                const int bh = bb * H_ + hh;
                const bf16 v = (bf16)acc[u][t][r];
                if (which == 0)      qb[((size_t)bh * T_ + tt) * D_ + dd] = v;
                else if (which == 1) kb[((size_t)bh * T_ + tt) * D_ + dd] = v;
                else                 vt[((size_t)bh * D_ + dd) * T_ + tt] = v;
            }
        }
    }
}

// ---------- Flash attention: one wave per 16-query tile, 32-key tiles, online softmax ----------
__global__ __launch_bounds__(128) void flash_attn_wmma(
    const bf16* __restrict__ qg,    // [B*H, T, D]
    const bf16* __restrict__ kg,    // [B*H, T, D]
    const bf16* __restrict__ vtg,   // [B*H, D, T]
    bf16* __restrict__ ao)          // [B, T, C] heads merged
{
    __shared__ __attribute__((aligned(64))) bf16 plds[4][16 * 32];  // per-wave P staging

    const int lane = threadIdx.x & 31;
    const int wid  = threadIdx.x >> 5;
    const int wave = blockIdx.x * 4 + wid;                  // 8192 waves
    const int qi = wave & 127;                              // T/16 query tiles
    const int bh = wave >> 7;
    const int q0 = qi * 16;
    const size_t base = (size_t)bh * T_ * D_;
    const int n = lane & 15, hf = lane >> 4;

    v16bf qlo = load_a16x32(qg + base + (size_t)q0 * D_, D_);
    v16bf qhi = load_a16x32(qg + base + (size_t)q0 * D_ + 32, D_);

    v8f o[4] = {vzero8(), vzero8(), vzero8(), vzero8()};
    float mrow[8], lrow[8];
#pragma unroll
    for (int r = 0; r < 8; ++r) { mrow[r] = -1e30f; lrow[r] = 0.f; }

    const int jcount = (q0 + 15) / 32 + 1;                  // causal tile bound

    // K^T B-fragments for tile 0 (B[d][key] = K[key][d], K-contiguous rows of [T,D])
    v16bf bk00 = load_b32x16(kg + base,                   D_);
    v16bf bk01 = load_b32x16(kg + base + 32,              D_);
    v16bf bk10 = load_b32x16(kg + base + (size_t)16 * D_,      D_);
    v16bf bk11 = load_b32x16(kg + base + (size_t)16 * D_ + 32, D_);

#pragma unroll 1
    for (int j = 0; j < jcount; ++j) {
        const int kbase = j * 32;
        // V B-fragments: in flight during score WMMAs + softmax
        v16bf bv0 = load_b32x16(vtg + base + (size_t)( 0) * T_ + kbase, T_);
        v16bf bv1 = load_b32x16(vtg + base + (size_t)(16) * T_ + kbase, T_);
        v16bf bv2 = load_b32x16(vtg + base + (size_t)(32) * T_ + kbase, T_);
        v16bf bv3 = load_b32x16(vtg + base + (size_t)(48) * T_ + kbase, T_);

        v8f s0 = vzero8(), s1 = vzero8();
        s0 = wmma_bf16(qlo, bk00, s0); s0 = wmma_bf16(qhi, bk01, s0);   // keys 0..15
        s1 = wmma_bf16(qlo, bk10, s1); s1 = wmma_bf16(qhi, bk11, s1);   // keys 16..31

        // pipeline: issue next tile's K loads, covered by softmax + P staging + PV WMMAs
        v16bf nk00 = bk00, nk01 = bk01, nk10 = bk10, nk11 = bk11;
        if (j + 1 < jcount) {
            const bf16* kp = kg + base + (size_t)(kbase + 32) * D_;
            nk00 = load_b32x16(kp,                    D_);
            nk01 = load_b32x16(kp + 32,               D_);
            nk10 = load_b32x16(kp + (size_t)16 * D_,      D_);
            nk11 = load_b32x16(kp + (size_t)16 * D_ + 32, D_);
        }

        float p0[8], p1[8], alpha[8];
#pragma unroll
        for (int r = 0; r < 8; ++r) {
            const int qrow = q0 + r + 8 * hf;               // this lane's row for VGPR r
            float a = s0[r] * 0.125f;                       // 1/sqrt(64)
            float b = s1[r] * 0.125f;
            if (kbase + n > qrow)      a = -1e30f;          // causal mask
            if (kbase + 16 + n > qrow) b = -1e30f;
            float t = fmaxf(a, b);                          // row-max over 16 lanes of half-wave
            t = fmaxf(t, __shfl_xor(t, 1));
            t = fmaxf(t, __shfl_xor(t, 2));
            t = fmaxf(t, __shfl_xor(t, 4));
            t = fmaxf(t, __shfl_xor(t, 8));
            const float mnew = fmaxf(mrow[r], t);
            alpha[r] = __expf(mrow[r] - mnew);
            mrow[r] = mnew;
            a = __expf(a - mnew);
            b = __expf(b - mnew);
            float s = a + b;
            s += __shfl_xor(s, 1);
            s += __shfl_xor(s, 2);
            s += __shfl_xor(s, 4);
            s += __shfl_xor(s, 8);
            lrow[r] = lrow[r] * alpha[r] + s;
            p0[r] = a; p1[r] = b;
        }

        // rescale O and stage P (C-layout -> row-major 16x32 in LDS)
        bf16* pw = &plds[wid][0];
#pragma unroll
        for (int r = 0; r < 8; ++r) {
            o[0][r] *= alpha[r]; o[1][r] *= alpha[r];
            o[2][r] *= alpha[r]; o[3][r] *= alpha[r];
            const int mr = r + 8 * hf;
            pw[mr * 32 + n]      = (bf16)p0[r];
            pw[mr * 32 + 16 + n] = (bf16)p1[r];
        }
        WAIT_DS();                                          // per-wave LDS RAW fence
        v16bf pf = load_a16x32(pw, 32);                     // A-fragment of P (16x32)

        o[0] = wmma_bf16(pf, bv0, o[0]);
        o[1] = wmma_bf16(pf, bv1, o[1]);
        o[2] = wmma_bf16(pf, bv2, o[2]);
        o[3] = wmma_bf16(pf, bv3, o[3]);

        bk00 = nk00; bk01 = nk01; bk10 = nk10; bk11 = nk11;
    }

    const int bb = bh >> 4, hh = bh & 15;
#pragma unroll
    for (int r = 0; r < 8; ++r) {
        const float inv = 1.0f / lrow[r];
        const int trow = q0 + r + 8 * hf;
        const size_t rowoff = ((size_t)(bb * T_ + trow)) * C_ + hh * D_;
#pragma unroll
        for (int t2 = 0; t2 < 4; ++t2)
            ao[rowoff + 16 * t2 + n] = (bf16)(o[t2][r] * inv);
    }
}

// ---------- Output projection: out = ao @ w_out^T + b_out (f32) ----------
__global__ __launch_bounds__(128) void out_gemm_wmma(
    const bf16* __restrict__ aob,   // [M, C]
    const bf16* __restrict__ wob,   // [C, C]
    const float* __restrict__ bias, // [C]
    float* __restrict__ out)        // [M, C] f32
{
    __shared__ __attribute__((aligned(128))) bf16 btile[2][64 * 32];
    const int lane = threadIdx.x & 31;
    const int wid  = threadIdx.x >> 5;
    const int blockRow0 = (blockIdx.x / 16) * 128;   // 64 x 16 blocks
    const int col0      = (blockIdx.x % 16) * 64;
    const int waveRow0  = blockRow0 + wid * 32;

    v8f acc[2][4] = {{vzero8(), vzero8(), vzero8(), vzero8()},
                     {vzero8(), vzero8(), vzero8(), vzero8()}};
    coop_gemm_mainloop(aob + (size_t)waveRow0 * C_, wob + (size_t)col0 * C_,
                       &btile[0][0], &btile[1][0], acc);

    const int n = lane & 15, hf = lane >> 4;
#pragma unroll
    for (int u = 0; u < 2; ++u) {
#pragma unroll
        for (int t = 0; t < 4; ++t) {
            const int gn = col0 + 16 * t + n;
            const float bv = bias[gn];
#pragma unroll
            for (int r = 0; r < 8; ++r) {
                const int gm = waveRow0 + 16 * u + r + 8 * hf;
                out[(size_t)gm * C_ + gn] = acc[u][t][r] + bv;
            }
        }
    }
}

// ---------- host launcher ----------
extern "C" void kernel_launch(void* const* d_in, const int* in_sizes, int n_in,
                              void* d_out, int out_size, void* d_ws, size_t ws_size,
                              hipStream_t stream) {
    (void)in_sizes; (void)n_in; (void)out_size; (void)ws_size;

    const float* x     = (const float*)d_in[0];
    const float* w_qkv = (const float*)d_in[1];
    const float* w_out = (const float*)d_in[2];
    const float* b_out = (const float*)d_in[3];
    float* out = (float*)d_out;

    char* ws = (char*)d_ws;
    size_t off = 0;
    auto carve = [&](size_t bytes) -> char* {
        char* p = ws + off;
        off += (bytes + 255) & ~(size_t)255;
        return p;
    };
    bf16* xb    = (bf16*)carve((size_t)M_ * C_ * 2);        // 16 MB
    bf16* wqkvb = (bf16*)carve((size_t)3 * C_ * C_ * 2);    //  6 MB
    bf16* woutb = (bf16*)carve((size_t)C_ * C_ * 2);        //  2 MB
    bf16* qb    = (bf16*)carve((size_t)M_ * C_ * 2);        // 16 MB
    bf16* kb    = (bf16*)carve((size_t)M_ * C_ * 2);        // 16 MB
    bf16* vt    = (bf16*)carve((size_t)M_ * C_ * 2);        // 16 MB
    bf16* ao    = (bf16*)carve((size_t)M_ * C_ * 2);        // 16 MB

    const int n1 = M_ * C_;
    cvt_f32_bf16<<<(n1 + 255) / 256, 256, 0, stream>>>(x, xb, n1);
    const int n2 = 3 * C_ * C_;
    cvt_f32_bf16<<<(n2 + 255) / 256, 256, 0, stream>>>(w_qkv, wqkvb, n2);
    const int n3 = C_ * C_;
    cvt_f32_bf16<<<(n3 + 255) / 256, 256, 0, stream>>>(w_out, woutb, n3);

    // QKV GEMM: (8192/128) x (3072/64) = 3072 blocks of 128 threads
    qkv_gemm_wmma<<<3072, 128, 0, stream>>>(xb, wqkvb, qb, kb, vt);
    // Attention: B*H*(T/16) = 8192 waves -> 2048 blocks
    flash_attn_wmma<<<2048, 128, 0, stream>>>(qb, kb, vt, ao);
    // Output GEMM: (8192/128) x (1024/64) = 1024 blocks
    out_gemm_wmma<<<1024, 128, 0, stream>>>(ao, woutb, b_out, out);
}
